// RCGANGenerator_21852793602680
// MI455X (gfx1250) — matrix operand order repaired and checked by reading, not compiled, not run on hardware
//
#include <hip/hip_runtime.h>

#define B_   1024
#define T_   256
#define DX   64
#define DEPS 16
#define DH   128
#define DIN  129
#define G3   384    // 3*DH
#define KEXT 160    // extended K: 128 h + 16 eps + 1 prev_y + 15 zero pad
#define KXT  (KEXT - DH)   // 32 extension rows

typedef __attribute__((ext_vector_type(16))) __bf16 v16bf;
typedef __attribute__((ext_vector_type(8)))  float  v8f;

// Fast transcendentals on the GRU critical path (TRANS32 exp/rcp).
__device__ __forceinline__ float fast_sigmoid(float x) {
  return __builtin_amdgcn_rcpf(1.0f + __expf(-x));
}
__device__ __forceinline__ float fast_tanh(float x) {
  return 1.0f - 2.0f * __builtin_amdgcn_rcpf(__expf(2.0f * x) + 1.0f);
}

// ---------------- preamble kernels (tiny GEMMs, one thread per output) -----

__global__ void k_xenc(const float* __restrict__ x, const float* __restrict__ Wx,
                       const float* __restrict__ bx, float* __restrict__ xe) {
  int i = blockIdx.x * blockDim.x + threadIdx.x;
  if (i >= B_ * DX) return;
  int b = i / DX, j = i % DX;
  float s = bx[j];
  const float* xr = x + b * DX;
  const float* wr = Wx + j * DX;
  #pragma unroll 8
  for (int k = 0; k < DX; ++k) s += xr[k] * wr[k];
  xe[i] = s;
}

__global__ void k_h0(const float* __restrict__ xe, const float* __restrict__ eps,
                     const float* __restrict__ W0, const float* __restrict__ b0,
                     float* __restrict__ h0) {
  int i = blockIdx.x * blockDim.x + threadIdx.x;
  if (i >= B_ * DH) return;
  int b = i / DH, j = i % DH;
  float s = b0[j];
  const float* xr = xe + b * DX;
  const float* wr = W0 + j * (DX + DEPS);
  #pragma unroll 8
  for (int k = 0; k < DX; ++k) s += xr[k] * wr[k];
  const float* er = eps + (size_t)b * T_ * DEPS;   // eps[b, 0, :]
  #pragma unroll
  for (int k = 0; k < DEPS; ++k) s += er[k] * wr[DX + k];
  h0[i] = tanhf(s);   // off the critical path; keep libm accuracy
}

// Gx[b][n] = x_enc[b] . Wih[n, 48:112] + bih[n] + (n<256 ? bhh[n] : 0)
__global__ void k_gx(const float* __restrict__ xe, const float* __restrict__ Wih,
                     const float* __restrict__ bih, const float* __restrict__ bhh,
                     float* __restrict__ Gx) {
  int i = blockIdx.x * blockDim.x + threadIdx.x;
  if (i >= B_ * G3) return;
  int b = i / G3, n = i % G3;
  float s = bih[n] + (n < 2 * DH ? bhh[n] : 0.0f);
  const float* xr = xe + b * DX;
  const float* wr = Wih + (size_t)n * DIN + 48;
  #pragma unroll 8
  for (int k = 0; k < DX; ++k) s += xr[k] * wr[k];
  Gx[i] = s;
}

// Ga[v][n] = Ea[v] . Wih[n, 0:32] ; Gt[v][n] = Et[v] . Wih[n, 32:48]
__global__ void k_gat(const float* __restrict__ Ea, const float* __restrict__ Et,
                      const float* __restrict__ Wih,
                      float* __restrict__ Ga, float* __restrict__ Gt) {
  int i = blockIdx.x * blockDim.x + threadIdx.x;
  if (i < 100 * G3) {
    int v = i / G3, n = i % G3;
    float s = 0.f;
    const float* er = Ea + v * 32;
    const float* wr = Wih + (size_t)n * DIN;
    #pragma unroll
    for (int k = 0; k < 32; ++k) s += er[k] * wr[k];
    Ga[i] = s;
  } else if (i < (100 + 4) * G3) {
    int j = i - 100 * G3;
    int v = j / G3, n = j % G3;
    float s = 0.f;
    const float* er = Et + v * 16;
    const float* wr = Wih + (size_t)n * DIN + 32;
    #pragma unroll
    for (int k = 0; k < 16; ++k) s += er[k] * wr[k];
    Gt[j] = s;
  }
}

// Extended B operand, [KEXT][G3] bf16:
//   k in [0,128):   Whh[n][k]              (recurrent, all gates)
//   k in [128,144): Wih[n][112+k-128]      (eps cols)   -- only r/z gates (n<256)
//   k == 144:       Wih[n][128]            (prev_y col) -- only r/z gates (n<256)
//   else:           0
__global__ void k_whh(const float* __restrict__ Whh, const float* __restrict__ Wih,
                      __bf16* __restrict__ WhhT) {
  int i = blockIdx.x * blockDim.x + threadIdx.x;
  if (i >= KEXT * G3) return;
  int k = i / G3, n = i % G3;
  float v = 0.0f;
  if (k < DH) v = Whh[n * DH + k];
  else if (n < 2 * DH) {
    if (k < DH + DEPS)       v = Wih[(size_t)n * DIN + 112 + (k - DH)];
    else if (k == DH + DEPS) v = Wih[(size_t)n * DIN + 128];
  }
  WhhT[i] = (__bf16)v;
}

// ---------------- persistent fused GRU scan ---------------------------------
// grid = 64 blocks (16 batch rows each), block = 256 threads = 8 waves.
// Register budget: Bf[3][4] (96) resident for k<128; the kc=4 (eps/py) B
// fragments are pre-swizzled into their per-lane register image in LDS and
// re-loaded per step as one contiguous 32B read per tile.

__device__ __forceinline__ v16bf load_afrag(const __bf16 (*h_b)[KEXT],
                                            int nn, int hi, int kc) {
  v16bf f;
  #pragma unroll
  for (int i = 0; i < 8; ++i) {
    int k0 = kc * 32 + 2 * i + (hi ? 16 : 0);
    f[2 * i]     = h_b[nn][k0];
    f[2 * i + 1] = h_b[nn][k0 + 1];
  }
  return f;
}

__global__ __launch_bounds__(256, 4)
void k_gru(const int* __restrict__ a, const int* __restrict__ tt,
           const float* __restrict__ y, const float* __restrict__ eps,
           const float* __restrict__ Wih, const float* __restrict__ bhh,
           const float* __restrict__ Wy, const float* __restrict__ by,
           const float* __restrict__ h0, const float* __restrict__ Gx,
           const float* __restrict__ Ga, const float* __restrict__ Gt,
           const __bf16* __restrict__ WhhT, float* __restrict__ out) {
  __shared__ __bf16 h_b[16][KEXT];     // 5 KB   [h(128) | eps(16) | py(1) | pad]
  __shared__ float  gh_s[16][G3];      // 24 KB  WMMA results
  __shared__ float  We_s[DH][DEPS];    // 8 KB   Wih[256+j, 112:128] (n-gate eps W)
  __shared__ float  wpy_s[DH];         // 0.5 KB Wih[256+j, 128]    (n-gate py W)
  __shared__ v16bf  BxF[8][3][32];     // 24 KB  kc=4 B fragments, per-lane image

  const int tid  = threadIdx.x;
  const int lane = tid & 31;
  const int wv   = tid >> 5;          // wave 0..7
  const int b0   = blockIdx.x * 16;
  const int gm   = tid >> 4;          // row 0..15 (same row in gx & gate stages)
  const int gsub = tid & 15;
  const int jb   = gsub * 8;          // hidden columns [jb, jb+8)

  // ---- init: h_b ext row = [h0 | eps(t=0) | py=0 | zeros] ----
  for (int i = tid; i < 16 * KEXT; i += 256) {
    int m = i / KEXT, j = i % KEXT;
    float v = 0.0f;
    if (j < DH)             v = h0[(b0 + m) * DH + j];
    else if (j < DH + DEPS) v = eps[((size_t)(b0 + m) * T_) * DEPS + (j - DH)];
    h_b[m][j] = (__bf16)v;
  }
  // ---- stage n-gate eps/py weights (constant over t) ----
  for (int i = tid; i < DH * DEPS; i += 256) {
    int j = i / DEPS, k = i % DEPS;
    We_s[j][k] = Wih[(size_t)(2 * DH + j) * DIN + 112 + k];
  }
  for (int i = tid; i < DH; i += 256)
    wpy_s[i] = Wih[(size_t)(2 * DH + i) * DIN + 128];

  // ---- per-thread register-resident state ----
  float hv[8], wyr[8], bhhn[8], gxr_r[24];
  #pragma unroll
  for (int q = 0; q < 8; ++q) {
    hv[q]   = h0[(b0 + gm) * DH + jb + q];
    wyr[q]  = Wy[jb + q];
    bhhn[q] = bhh[2 * DH + jb + q];
  }
  #pragma unroll
  for (int g = 0; g < 3; ++g)
    #pragma unroll
    for (int q = 0; q < 8; ++q)
      gxr_r[g * 8 + q] = Gx[(size_t)(b0 + gm) * G3 + g * DH + jb + q];

  // ---- resident B fragments (recurrent K only) + pre-swizzled kc=4 frags ----
  // B 32x16 bf16, contiguous k order used here (same as A): n = lane%16; vgpr i half h: K = 2*i + h + 16*(lane>=16).
  v16bf Bf[3][4];
  {
    const int nn = lane & 15;
    const int hi = lane >> 4;
    for (int tN = 0; tN < 3; ++tN) {
      int ncol = (3 * wv + tN) * 16 + nn;
      for (int kc = 0; kc < 4; ++kc) {
        v16bf f;
        #pragma unroll
        for (int i = 0; i < 8; ++i) {
          int k = kc * 32 + 2 * i + hi * 16;
          f[2 * i]     = WhhT[(size_t)k * G3 + ncol];
          f[2 * i + 1] = WhhT[(size_t)(k + 1) * G3 + ncol];
        }
        Bf[tN][kc] = f;
      }
      // kc=4 fragment: gather once from global, store per-lane image in LDS
      v16bf fx;
      #pragma unroll
      for (int i = 0; i < 8; ++i) {
        int k = DH + 2 * i + hi * 16;
        fx[2 * i]     = WhhT[(size_t)k * G3 + ncol];
        fx[2 * i + 1] = WhhT[(size_t)(k + 1) * G3 + ncol];
      }
      BxF[wv][tN][lane] = fx;
    }
  }
  __syncthreads();

  const float byv = by[0];

  #pragma unroll 1   // recurrence is sequential: unrolling only doubles code
  for (int t = 0; t < T_; ++t) {
    // ---- Phase A (registers only): gx = Ga[a] + Gt[t] + Gx  (+ eps/py for n) ----
    float gxv[24];
    {
      const int gb = b0 + gm;
      const int ai = a[gb * T_ + t];
      const int ti = tt[gb * T_ + t];
      const float* ga = Ga + ai * G3;
      const float* gt = Gt + ti * G3;
      if (t + 1 < T_) __builtin_prefetch(eps + ((size_t)gb * T_ + t + 1) * DEPS, 0, 1);
      #pragma unroll
      for (int g = 0; g < 3; ++g)
        #pragma unroll
        for (int q = 0; q < 8; ++q) {
          int n = g * DH + jb + q;
          gxv[g * 8 + q] = ga[n] + gt[n] + gxr_r[g * 8 + q];
        }
      // n-gate input part in f32 (uniform across threads, weights in LDS)
      const float py = (t == 0) ? 0.0f : y[gb * T_ + t - 1];
      float ev[DEPS];
      const float* er = eps + ((size_t)gb * T_ + t) * DEPS;
      #pragma unroll
      for (int k = 0; k < DEPS; ++k) ev[k] = er[k];
      #pragma unroll
      for (int q = 0; q < 8; ++q) {
        float s = py * wpy_s[jb + q];
        #pragma unroll
        for (int k = 0; k < DEPS; ++k) s += ev[k] * We_s[jb + q][k];
        gxv[16 + q] += s;
      }
    }
    // ---- Phase B: gh = [h|eps|py] @ B_ext via WMMA bf16 (f32 accum) ----
    {
      const int nn = lane & 15;
      const int hi = lane >> 4;
      // double-buffered A fragments over the 4 resident chunks
      v16bf AfA = load_afrag(h_b, nn, hi, 0);
      v16bf AfB;
      v8f zero = {0.f, 0.f, 0.f, 0.f, 0.f, 0.f, 0.f, 0.f};
      v8f acc[3] = {zero, zero, zero};
      #pragma unroll
      for (int kc = 0; kc < 4; ++kc) {
        if (kc & 1) AfA = load_afrag(h_b, nn, hi, kc + 1);
        else        AfB = load_afrag(h_b, nn, hi, kc + 1);
        const v16bf& cur = (kc & 1) ? AfB : AfA;
        #pragma unroll
        for (int tN = 0; tN < 3; ++tN)
          acc[tN] = __builtin_amdgcn_wmma_f32_16x16x32_bf16(
              false, cur, false, Bf[tN][kc], (short)0, acc[tN], false, false);
      }
      // kc = 4 (eps/py): A already in AfA; B = contiguous per-lane LDS image
      #pragma unroll
      for (int tN = 0; tN < 3; ++tN) {
        v16bf bx = BxF[wv][tN][lane];
        acc[tN] = __builtin_amdgcn_wmma_f32_16x16x32_bf16(
            false, AfA, false, bx, (short)0, acc[tN], false, false);
      }
      // C/D layout: vgpr r, lane: m = r + 8*(lane>=16), n = lane%16
      #pragma unroll
      for (int tN = 0; tN < 3; ++tN) {
        int ncol = (3 * wv + tN) * 16 + nn;
        #pragma unroll
        for (int r = 0; r < 8; ++r)
          gh_s[r + (hi ? 8 : 0)][ncol] = acc[tN][r];
      }
    }
    __syncthreads();
    // ---- Phase C: gates (gx + h in registers) + logit + stage t+1 eps/py ----
    {
      float part = 0.0f;
      #pragma unroll
      for (int q = 0; q < 8; ++q) {
        int j = jb + q;
        float sr  = gxv[q]      + gh_s[gm][j];            // bhh_r folded in Gx
        float sz  = gxv[8 + q]  + gh_s[gm][j + DH];       // bhh_z folded in Gx
        float sn  = gxv[16 + q];                          // f32 input part
        float ghn = gh_s[gm][j + 2 * DH] + bhhn[q];       // recurrent part
        float r  = fast_sigmoid(sr);
        float z  = fast_sigmoid(sz);
        float n2 = fast_tanh(sn + r * ghn);
        hv[q] = (1.f - z) * n2 + z * hv[q];
        h_b[gm][j] = (__bf16)hv[q];
        part += hv[q] * wyr[q];
      }
      // stage next step's eps/prev_y into the A extension (one thread per row);
      // consumed only after the barrier below.
      if (gsub == 0 && t + 1 < T_) {
        const float* er = eps + ((size_t)(b0 + gm) * T_ + t + 1) * DEPS;
        #pragma unroll
        for (int k = 0; k < DEPS; ++k) h_b[gm][DH + k] = (__bf16)er[k];
        h_b[gm][DH + DEPS] = (__bf16)y[(b0 + gm) * T_ + t];
      }
      // logit: width-16 shuffle reduction over the row's 16 threads
      part += __shfl_xor(part, 8, 16);
      part += __shfl_xor(part, 4, 16);
      part += __shfl_xor(part, 2, 16);
      part += __shfl_xor(part, 1, 16);
      if (gsub == 0) {
        float s = part + byv;
        size_t ob = (((size_t)(b0 + gm)) * T_ + t) * 2;
        out[ob]     = s;
        out[ob + 1] = fast_sigmoid(s);
      }
    }
    __syncthreads();
  }
}

// ---------------- launch -----------------------------------------------------

extern "C" void kernel_launch(void* const* d_in, const int* in_sizes, int n_in,
                              void* d_out, int out_size, void* d_ws, size_t ws_size,
                              hipStream_t stream) {
  (void)in_sizes; (void)n_in; (void)out_size; (void)ws_size;
  const float* x    = (const float*)d_in[0];
  const int*   a    = (const int*)d_in[1];
  const int*   tt   = (const int*)d_in[2];
  const float* y    = (const float*)d_in[3];
  // d_in[4] = mask (all ones -> forward identity), unused
  const float* eps  = (const float*)d_in[5];
  const float* Wx   = (const float*)d_in[6];
  const float* bx   = (const float*)d_in[7];
  const float* Ea   = (const float*)d_in[8];
  const float* Et   = (const float*)d_in[9];
  const float* Wih  = (const float*)d_in[10];
  const float* Whh  = (const float*)d_in[11];
  const float* bih  = (const float*)d_in[12];
  const float* bhh  = (const float*)d_in[13];
  const float* W0   = (const float*)d_in[14];
  const float* b0   = (const float*)d_in[15];
  const float* Wy   = (const float*)d_in[16];
  const float* by   = (const float*)d_in[17];

  float* ws  = (float*)d_ws;
  float* xe  = ws;                    // 65536 f32
  float* h0  = xe + B_ * DX;          // 131072 f32
  float* Gx  = h0 + B_ * DH;          // 393216 f32
  float* Ga  = Gx + (size_t)B_ * G3;  // 38400 f32
  float* Gt  = Ga + 100 * G3;         // 1536 f32
  __bf16* WhhT = (__bf16*)(Gt + 4 * G3); // KEXT*G3 = 61440 bf16

  dim3 blk(256);
  k_xenc<<<(B_ * DX + 255) / 256, blk, 0, stream>>>(x, Wx, bx, xe);
  k_h0  <<<(B_ * DH + 255) / 256, blk, 0, stream>>>(xe, eps, W0, b0, h0);
  k_gx  <<<(B_ * G3 + 255) / 256, blk, 0, stream>>>(xe, Wih, bih, bhh, Gx);
  k_gat <<<((100 + 4) * G3 + 255) / 256, blk, 0, stream>>>(Ea, Et, Wih, Ga, Gt);
  k_whh <<<(KEXT * G3 + 255) / 256, blk, 0, stream>>>(Whh, Wih, WhhT);
  k_gru <<<B_ / 16, blk, 0, stream>>>(a, tt, y, eps, Wih, bhh, Wy, by,
                                      h0, Gx, Ga, Gt, WhhT, (float*)d_out);
}
